// SelfAttention_24601572671941
// MI455X (gfx1250) — matrix operand order repaired and checked by reading, not compiled, not run on hardware
//
#include <hip/hip_runtime.h>

// ---------------------------------------------------------------------------
// Self-attention forward for MI455X (gfx1250, wave32, WMMA f16 -> f32 acc)
// B=2, T=2048, C=1024, H=16, DH=64
// ---------------------------------------------------------------------------

typedef _Float16 h8  __attribute__((ext_vector_type(8)));
typedef _Float16 h16 __attribute__((ext_vector_type(16)));
typedef float    f8  __attribute__((ext_vector_type(8)));

#define T_LEN  2048
#define C_DIM  1024
#define H_NUM  16
#define DH_DIM 64

static __device__ __forceinline__ h16 mk16(h8 lo, h8 hi) {
    return __builtin_shufflevector(lo, hi, 0,1,2,3,4,5,6,7,8,9,10,11,12,13,14,15);
}

static __device__ __forceinline__ f8 wmma_f16(h16 a, h16 b, f8 c) {
    // D = A(16x32) * B(32x16) + C, f32 accumulate
    return __builtin_amdgcn_wmma_f32_16x16x32_f16(false, a, false, b, (short)0, c, false, false);
}

// ---------------------------------------------------------------------------
// Weight transpose + f32 -> f16 convert: Wt[n][k] = (f16) W[k][n]  (1024x1024)
// ---------------------------------------------------------------------------
__global__ __launch_bounds__(256) void k_wt(const float* __restrict__ W,
                                            _Float16* __restrict__ Wt) {
    int id = blockIdx.x * 256 + threadIdx.x;     // 0 .. 1M-1
    int n = id >> 10;
    int k = id & 1023;
    Wt[id] = (_Float16)W[k * 1024 + n];
}

// ---------------------------------------------------------------------------
// WMMA GEMM: C[4096x1024] = A[4096x1024] * Wt^T  (Wt is N-major f16)
// Block tile 128x128, K-step 32, 8 waves; wave tile 32x64 (2x4 WMMA tiles).
// MODE 0: out f16, head-major [bh][t][d], scale 1/8   (Q)
// MODE 1: out f16, head-major [bh][t][d]              (K)
// MODE 2: out f16, transposed [bh][d][t]              (V^T)
// MODE 3: A is f16, out f32 row-major + bias          (y @ Wo + bo)
// ---------------------------------------------------------------------------
template <int MODE>
__global__ __launch_bounds__(256) void k_gemm(const float* __restrict__ A32,
                                              const _Float16* __restrict__ A16,
                                              const _Float16* __restrict__ Bt,
                                              _Float16* __restrict__ O16,
                                              float* __restrict__ O32,
                                              const float* __restrict__ bias) {
    __shared__ _Float16 As[128 * 40];   // 128 rows (M), 32 K, stride 40
    __shared__ _Float16 Bs[128 * 40];   // 128 rows (N), 32 K, stride 40

    const int tid  = threadIdx.x;
    const int m0   = blockIdx.y * 128;
    const int n0   = blockIdx.x * 128;
    const int w    = tid >> 5;
    const int ln   = tid & 31;
    const int l16  = ln & 15;
    const int hl   = ln >> 4;          // lane half (0/1)
    const int wm   = w >> 1;           // 0..3
    const int wn   = w & 1;            // 0..1

    f8 acc[2][4];
#pragma unroll
    for (int a = 0; a < 2; ++a)
#pragma unroll
        for (int b = 0; b < 4; ++b)
#pragma unroll
            for (int j = 0; j < 8; ++j) acc[a][b][j] = 0.0f;

    const int r  = tid >> 1;           // staging row 0..127
    const int ch = (tid & 1) * 16;     // staging K-offset 0/16

    for (int kb = 0; kb < C_DIM; kb += 32) {
        // ---- stage A tile ----
        if (MODE == 3) {
            const _Float16* src = A16 + (size_t)(m0 + r) * C_DIM + kb + ch;
            *(h8*)&As[r * 40 + ch]     = *(const h8*)src;
            *(h8*)&As[r * 40 + ch + 8] = *(const h8*)(src + 8);
        } else {
            const float* src = A32 + (size_t)(m0 + r) * C_DIM + kb + ch;
            h8 v0, v1;
#pragma unroll
            for (int j = 0; j < 8; ++j) { v0[j] = (_Float16)src[j]; v1[j] = (_Float16)src[8 + j]; }
            *(h8*)&As[r * 40 + ch]     = v0;
            *(h8*)&As[r * 40 + ch + 8] = v1;
        }
        // ---- stage B tile (Wt is N-major f16) ----
        {
            const _Float16* src = Bt + (size_t)(n0 + r) * C_DIM + kb + ch;
            *(h8*)&Bs[r * 40 + ch]     = *(const h8*)src;
            *(h8*)&Bs[r * 40 + ch + 8] = *(const h8*)(src + 8);
        }
        __syncthreads();

        // ---- fragments ----
        h16 af[2], bf[4];
#pragma unroll
        for (int mt = 0; mt < 2; ++mt) {
            int row = wm * 32 + mt * 16 + l16;
            // A layout: lane holds 16 contiguous K starting at 16*hl
            af[mt] = mk16(*(const h8*)&As[row * 40 + 16 * hl],
                          *(const h8*)&As[row * 40 + 16 * hl + 8]);
        }
#pragma unroll
        for (int nt = 0; nt < 4; ++nt) {
            int nr = wn * 64 + nt * 16 + l16;
            // B layout: lane holds 16 contiguous K starting at 16*hl
            bf[nt] = mk16(*(const h8*)&Bs[nr * 40 + 16 * hl],
                          *(const h8*)&Bs[nr * 40 + 16 * hl + 8]);
        }
#pragma unroll
        for (int mt = 0; mt < 2; ++mt)
#pragma unroll
            for (int nt = 0; nt < 4; ++nt)
                acc[mt][nt] = wmma_f16(af[mt], bf[nt], acc[mt][nt]);
        __syncthreads();
    }

    // ---- epilogue ----
#pragma unroll
    for (int mt = 0; mt < 2; ++mt) {
#pragma unroll
        for (int nt = 0; nt < 4; ++nt) {
#pragma unroll
            for (int v = 0; v < 8; ++v) {
                int m = m0 + wm * 32 + mt * 16 + v + 8 * hl;
                int n = n0 + wn * 64 + nt * 16 + l16;
                float val = acc[mt][nt][v];
                if (MODE == 3) {
                    O32[(size_t)m * C_DIM + n] = val + bias[n];
                } else {
                    int b = m >> 11, t = m & 2047;
                    int hh = n >> 6, d = n & 63;
                    int bh = b * H_NUM + hh;
                    if (MODE == 0) val *= 0.125f;   // DH^-0.5
                    if (MODE == 2)
                        O16[((size_t)bh * DH_DIM + d) * T_LEN + t] = (_Float16)val;
                    else
                        O16[((size_t)bh * T_LEN + t) * DH_DIM + d] = (_Float16)val;
                }
            }
        }
    }
}

// ---------------------------------------------------------------------------
// Flash attention, register-only. Each wave owns 16 query rows.
// Computes S^T = K * Q^T per 32-key step (2 WMMA-tile columns), online
// softmax with a single shfl_xor(16) per reduction, repacks exp'd P^T into
// the B-fragment layout with 16 shuffles, accumulates O^T = V^T * P^T.
// ---------------------------------------------------------------------------
__global__ __launch_bounds__(256) void k_attn(const _Float16* __restrict__ Q,
                                              const _Float16* __restrict__ Kh,
                                              const _Float16* __restrict__ Vt,
                                              const float* __restrict__ rel,
                                              _Float16* __restrict__ Y) {
    const int tid = threadIdx.x;
    const int w   = tid >> 5;
    const int ln  = tid & 31;
    const int nq  = ln & 15;           // q-row within wave tile / column lane
    const int hl  = ln >> 4;           // lane half

    const int hh = blockIdx.y;
    const int b  = blockIdx.z;
    const int bh = b * H_NUM + hh;
    const int q0 = blockIdx.x * 128 + w * 16;  // first q row of this wave
    const int qi = q0 + nq;                    // this lane's q row (0..2047)

    const _Float16* qp = Q  + (size_t)bh * T_LEN * DH_DIM;
    const _Float16* kp = Kh + (size_t)bh * T_LEN * DH_DIM;
    const _Float16* vp = Vt + (size_t)bh * DH_DIM * T_LEN;
    const float* relp  = rel + (size_t)hh * (2 * T_LEN - 1) + (T_LEN - 1);

    // Q as B-operand fragments (d chunks of 32): lane nq holds row qi,
    // 16 contiguous d starting at 32*f + 16*hl.
    h16 qf[2];
#pragma unroll
    for (int f = 0; f < 2; ++f) {
        const _Float16* p = qp + (size_t)qi * DH_DIM + 32 * f + 16 * hl;
        qf[f] = mk16(*(const h8*)p, *(const h8*)(p + 8));
    }

    f8 oacc[4];
#pragma unroll
    for (int dt = 0; dt < 4; ++dt)
#pragma unroll
        for (int j = 0; j < 8; ++j) oacc[dt][j] = 0.0f;

    float rmax = -1e30f;
    float rsum = 0.0f;

    for (int jb = 0; jb <= q0 + 15; jb += 32) {
        // ---- S^T = K * Q^T : two 16x16 tiles (keys jb..+15, jb+16..+31) ----
        f8 s[2];
#pragma unroll
        for (int t2 = 0; t2 < 2; ++t2) {
            int key = jb + 16 * t2 + nq;   // A-matrix row = key (lane nq)
            f8 a;
#pragma unroll
            for (int j = 0; j < 8; ++j) a[j] = 0.0f;
#pragma unroll
            for (int f = 0; f < 2; ++f) {
                const _Float16* p = kp + (size_t)key * DH_DIM + 32 * f;
                h16 kf = mk16(*(const h8*)(p + 16 * hl),
                              *(const h8*)(p + 16 * hl + 8));
                a = wmma_f16(kf, qf[f], a);
            }
            s[t2] = a;
        }

        // ---- relative bias + causal mask (S^T: lane=qrow, vgpr=key) ----
        float mx = -1e30f;
#pragma unroll
        for (int t2 = 0; t2 < 2; ++t2) {
#pragma unroll
            for (int v = 0; v < 8; ++v) {
                int jc = jb + 16 * t2 + v + 8 * hl;       // key index
                float sv = s[t2][v] + relp[qi - jc];      // bias[h][i-j+T-1]
                if (jc > qi) sv = -1e30f;                 // causal
                s[t2][v] = sv;
                mx = fmaxf(mx, sv);
            }
        }
        mx = fmaxf(mx, __shfl_xor(mx, 16, 32));           // full-row max
        float nmax  = fmaxf(rmax, mx);
        float alpha = __expf(rmax - nmax);
        rmax = nmax;

        float lsum = 0.0f;
#pragma unroll
        for (int t2 = 0; t2 < 2; ++t2)
#pragma unroll
            for (int v = 0; v < 8; ++v) {
                float pv = __expf(s[t2][v] - nmax);
                s[t2][v] = pv;
                lsum += pv;
            }
        lsum += __shfl_xor(lsum, 16, 32);                 // full-row sum
        rsum = rsum * alpha + lsum;
#pragma unroll
        for (int dt = 0; dt < 4; ++dt)
#pragma unroll
            for (int j = 0; j < 8; ++j) oacc[dt][j] *= alpha;

        // ---- repack P^T (f32 acc layout) -> B-fragment (h16) ----
        // B-frag lane l needs keys 16*hl+0..7 from lane nq (half 0 keys come
        // from e-values whose acc-half matches) and keys 16*hl+8..15 from
        // lane nq+16.  One xor-16 exchange per element.
        h16 pf;
#pragma unroll
        for (int kk = 0; kk < 8; ++kk) {
            float e0 = s[0][kk];
            float e1 = s[1][kk];
            float x0 = __shfl_xor(e0, 16, 32);
            float x1 = __shfl_xor(e1, 16, 32);
            float lo = (hl == 0) ? e0 : x1;   // keys 16*hl + 0..7  (src lane nq)
            float hi = (hl == 0) ? x0 : e1;   // keys 16*hl + 8..15 (src lane nq+16)
            pf[kk]     = (_Float16)lo;
            pf[kk + 8] = (_Float16)hi;
        }

        // ---- O^T += V^T * P^T : 4 d-tiles of 16 ----
#pragma unroll
        for (int dt = 0; dt < 4; ++dt) {
            const _Float16* p = vp + (size_t)(16 * dt + nq) * T_LEN + jb;
            h16 vf = mk16(*(const h8*)(p + 16 * hl),
                          *(const h8*)(p + 16 * hl + 8));
            oacc[dt] = wmma_f16(vf, pf, oacc[dt]);
        }
    }

    // ---- normalize and write y[b*T+t][h*64+d] (f16, row-major for pass 3) --
    float inv = 1.0f / rsum;
#pragma unroll
    for (int dt = 0; dt < 4; ++dt) {
#pragma unroll
        for (int v = 0; v < 8; ++v) {
            int d = hh * DH_DIM + 16 * dt + v + 8 * hl;
            Y[((size_t)(b * T_LEN + qi)) * C_DIM + d] = (_Float16)(oacc[dt][v] * inv);
        }
    }
}

// ---------------------------------------------------------------------------
// Launch
// ---------------------------------------------------------------------------
extern "C" void kernel_launch(void* const* d_in, const int* in_sizes, int n_in,
                              void* d_out, int out_size, void* d_ws, size_t ws_size,
                              hipStream_t stream) {
    (void)in_sizes; (void)n_in; (void)out_size; (void)ws_size;

    const float* x   = (const float*)d_in[0];
    const float* Wq  = (const float*)d_in[1];
    const float* Wk  = (const float*)d_in[2];
    const float* Wv  = (const float*)d_in[3];
    const float* Wo  = (const float*)d_in[4];
    const float* bo  = (const float*)d_in[5];
    const float* rel = (const float*)d_in[6];
    // d_in[7]: causal mask — implicit in the attention kernel

    // workspace layout (f16): 4x weightT (2MB) + Q/K/Vt (8MB each) + y (8MB) = 40MB
    _Float16* wtq = (_Float16*)d_ws;
    _Float16* wtk = wtq + (1u << 20);
    _Float16* wtv = wtk + (1u << 20);
    _Float16* wto = wtv + (1u << 20);
    _Float16* qb  = wto + (1u << 20);
    _Float16* kbf = qb  + (4u << 20);
    _Float16* vtb = kbf + (4u << 20);
    _Float16* yb  = vtb + (4u << 20);
    float* out = (float*)d_out;

    k_wt<<<4096, 256, 0, stream>>>(Wq, wtq);
    k_wt<<<4096, 256, 0, stream>>>(Wk, wtk);
    k_wt<<<4096, 256, 0, stream>>>(Wv, wtv);
    k_wt<<<4096, 256, 0, stream>>>(Wo, wto);

    dim3 gg(8, 32);   // N/128, M/128
    k_gemm<0><<<gg, 256, 0, stream>>>(x, nullptr, wtq, qb,  nullptr, nullptr);
    k_gemm<1><<<gg, 256, 0, stream>>>(x, nullptr, wtk, kbf, nullptr, nullptr);
    k_gemm<2><<<gg, 256, 0, stream>>>(x, nullptr, wtv, vtb, nullptr, nullptr);

    k_attn<<<dim3(16, 16, 2), 256, 0, stream>>>(qb, kbf, vtb, rel, yb);

    k_gemm<3><<<gg, 256, 0, stream>>>(nullptr, yb, wto, nullptr, out, bo);
}
